// MoELayer_84705345012275
// MI455X (gfx1250) — compile-verified, matrix operand order checked by reading, not run on hardware
//
#include <hip/hip_runtime.h>

// ---------------------------------------------------------------------------
// Types for CDNA5 WMMA (wave32, 16x16x32 bf16 -> f32)
// ---------------------------------------------------------------------------
typedef __bf16 v16bf __attribute__((ext_vector_type(16)));
typedef float  v8f   __attribute__((ext_vector_type(8)));

union FragBF {
    v16bf v;
    unsigned short us[16];
    unsigned int   u[8];
};

__device__ __forceinline__ unsigned short f2bf_bits(float f) {
    unsigned u = __float_as_uint(f);
    unsigned r = u + 0x7fffu + ((u >> 16) & 1u);   // round-to-nearest-even
    return (unsigned short)(r >> 16);
}

__device__ __forceinline__ v8f wmma_bf16(const FragBF& a, const FragBF& b, v8f c) {
    return __builtin_amdgcn_wmma_f32_16x16x32_bf16(
        /*neg_a=*/false, a.v, /*neg_b=*/false, b.v,
        /*c_mod=*/(short)0, c, /*reuse_a=*/false, /*reuse_b=*/false);
}

// Packed-fragment offset: element (maj, k) of a [Maj x K] operand, where maj
// is the dimension striped across lanes (M for A, N for B). Fragment for
// (majtile, kstep) = 32 lanes x 16 bf16, contiguous 1KB block.
__device__ __forceinline__ size_t pack_off(int maj, int k, int K) {
    int tile   = maj >> 4;
    int kstep  = k >> 5;
    int kloc   = k & 31;
    int group  = kloc >> 4;          // which 16-wide K group
    int rem    = kloc & 15;
    int half   = rem >> 3;           // lane half (lanes 0-15 vs 16-31)
    int within = rem & 7;
    int lane   = (half << 4) + (maj & 15);
    int idx    = ((group << 2) + (within >> 1)) * 2 + (within & 1);  // == 2v+elem
    return ((size_t)(tile * (K >> 5) + kstep) * 32 + lane) * 16 + idx;
}

// ---------------------------------------------------------------------------
// Problem constants
// ---------------------------------------------------------------------------
#define Bsz 8
#define Lseq 512
#define Dm 512
#define Hn 8
#define DH 64
#define Rows 4096              // B*L
#define NElem 2097152          // B*L*D

// ---------------------------------------------------------------------------
// f32 row-major [K,N] weight -> packed bf16 B-fragments
// ---------------------------------------------------------------------------
__global__ __launch_bounds__(256) void cvt_pack_kernel(const float* __restrict__ in,
                                                       unsigned short* __restrict__ out,
                                                       int K, int N) {
    int i = blockIdx.x * 256 + threadIdx.x;
    if (i >= K * N) return;
    int k = i / N, n = i - k * N;
    out[pack_off(n, k, K)] = f2bf_bits(in[i]);
}

// ---------------------------------------------------------------------------
// LayerNorm rows of 512, f32 in -> packed bf16 A-fragments (K = 512)
// ---------------------------------------------------------------------------
__global__ __launch_bounds__(256) void ln_kernel(const float* __restrict__ x,
                                                 const float* __restrict__ s,
                                                 const float* __restrict__ b,
                                                 unsigned short* __restrict__ out) {
    __shared__ float red[256], red2[256];
    int row = blockIdx.x, t = threadIdx.x;
    const float* xr = x + (size_t)row * Dm;
    float v0 = xr[t], v1 = xr[t + 256];
    red[t] = v0 + v1; red2[t] = v0 * v0 + v1 * v1;
    __syncthreads();
    for (int k = 128; k > 0; k >>= 1) {
        if (t < k) { red[t] += red[t + k]; red2[t] += red2[t + k]; }
        __syncthreads();
    }
    float m   = red[0] * (1.f / Dm);
    float var = red2[0] * (1.f / Dm) - m * m;
    float inv = rsqrtf(var + 1e-5f);
    out[pack_off(row, t, Dm)]       = f2bf_bits((v0 - m) * inv * s[t] + b[t]);
    out[pack_off(row, t + 256, Dm)] = f2bf_bits((v1 - m) * inv * s[t + 256] + b[t + 256]);
}

// ---------------------------------------------------------------------------
// WMMA bf16 GEMM on packed fragments: C[M,N] = A*B + bias, epilogues.
// One wave computes a 16x64 output strip (1 A-frag reused for 4 B-frags).
// ---------------------------------------------------------------------------
#define MODE_ROW_BF16  0   // store bf16 row-major (qkv for attention)
#define MODE_RESID     1   // + resid, store f32 row-major
#define MODE_GELU_PACK 2   // gelu, store packed bf16 A-fragments (K=N)

__global__ __launch_bounds__(256)
void gemm_bf16(const FragBF* __restrict__ Af, const FragBF* __restrict__ Bf,
               const float* __restrict__ bias, const float* __restrict__ resid,
               void* __restrict__ outp, int M, int N, int K, int mode) {
    int wave = threadIdx.x >> 5;
    int lane = threadIdx.x & 31;
    int ksteps = K >> 5;
    int nquads = N >> 6;
    int t = blockIdx.x * 8 + wave;
    if (t >= (M >> 4) * nquads) return;          // wave-uniform exit
    int tm = t / nquads, tq = t % nquads;
    int half = lane >> 4, l15 = lane & 15;

    const FragBF* Ap  = Af + (size_t)tm * ksteps * 32 + lane;
    const FragBF* Bp0 = Bf + (size_t)(tq * 4 + 0) * ksteps * 32 + lane;
    const FragBF* Bp1 = Bf + (size_t)(tq * 4 + 1) * ksteps * 32 + lane;
    const FragBF* Bp2 = Bf + (size_t)(tq * 4 + 2) * ksteps * 32 + lane;
    const FragBF* Bp3 = Bf + (size_t)(tq * 4 + 3) * ksteps * 32 + lane;

    v8f c0 = {}, c1 = {}, c2 = {}, c3 = {};
    for (int ks = 0; ks < ksteps; ++ks) {
        FragBF a  = Ap[(size_t)ks * 32];         // 2 x global_load_b128
        FragBF b0 = Bp0[(size_t)ks * 32];
        FragBF b1 = Bp1[(size_t)ks * 32];
        FragBF b2 = Bp2[(size_t)ks * 32];
        FragBF b3 = Bp3[(size_t)ks * 32];
        if (ks + 1 < ksteps) {                   // gfx1250 global_prefetch path
            __builtin_prefetch((const void*)(Ap  + (size_t)(ks + 1) * 32), 0, 1);
            __builtin_prefetch((const void*)(Bp0 + (size_t)(ks + 1) * 32), 0, 1);
        }
        c0 = wmma_bf16(a, b0, c0);
        c1 = wmma_bf16(a, b1, c1);
        c2 = wmma_bf16(a, b2, c2);
        c3 = wmma_bf16(a, b3, c3);
    }

#pragma unroll
    for (int i = 0; i < 4; ++i) {
        v8f c = (i == 0) ? c0 : (i == 1) ? c1 : (i == 2) ? c2 : c3;
        int col = (tq * 4 + i) * 16 + l15;
        float bv = bias ? bias[col] : 0.f;
#pragma unroll
        for (int r = 0; r < 8; ++r) {
            int row = tm * 16 + r + half * 8;
            float val = c[r] + bv;
            if (mode == MODE_RESID) {
                size_t idx = (size_t)row * N + col;
                ((float*)outp)[idx] = val + resid[idx];
            } else if (mode == MODE_GELU_PACK) {
                float x3 = val * val * val;
                float g = 0.5f * val *
                          (1.f + tanhf(0.7978845608028654f * (val + 0.044715f * x3)));
                ((unsigned short*)outp)[pack_off(row, col, N)] = f2bf_bits(g);
            } else {
                ((unsigned short*)outp)[(size_t)row * N + col] = f2bf_bits(val);
            }
        }
    }
}

// ---------------------------------------------------------------------------
// Fused attention: one block per (b, h, 16-row q-tile). 4 waves.
// Scores [16,512] in LDS; WMMA for Q*K^T and P*V.
// qkv: bf16 row-major [B, L, 3D]; out: packed bf16 A-fragments (K = 512)
// ---------------------------------------------------------------------------
__global__ __launch_bounds__(128)
void attn_kernel(const unsigned short* __restrict__ qkv, unsigned short* __restrict__ out) {
    __shared__ float sc[16][Lseq];   // 32 KB
    __shared__ float part[16][8];

    int blk = blockIdx.x;
    int qt  = blk & 31;              // L/16 q tiles
    int h   = (blk >> 5) & 7;
    int b   = blk >> 8;
    int wave = threadIdx.x >> 5, lane = threadIdx.x & 31;
    int half = lane >> 4, l15 = lane & 15;
    const int D3 = 3 * Dm;
    const unsigned short* base = qkv + (size_t)b * Lseq * D3;

    // ---- Stage 1: scores = Q * K^T / 8 ----
    for (int tn = wave; tn < 32; tn += 4) {
        v8f c = {};
        for (int k0 = 0; k0 < DH; k0 += 32) {
            FragBF a, bm;
#pragma unroll
            for (int v = 0; v < 8; ++v) {
                int kk = k0 + ((v >> 2) << 4) + (half << 3) + ((v & 3) << 1);
                a.u[v]  = *(const unsigned int*)(base + (size_t)(qt * 16 + l15) * D3 + h * DH + kk);
                bm.u[v] = *(const unsigned int*)(base + (size_t)(tn * 16 + l15) * D3 + Dm + h * DH + kk);
            }
            c = wmma_bf16(a, bm, c);
        }
#pragma unroll
        for (int r = 0; r < 8; ++r)
            sc[r + half * 8][tn * 16 + l15] = c[r] * 0.125f;   // 1/sqrt(64)
    }
    __syncthreads();

    // ---- Stage 2: row softmax (8 threads per row) ----
    int r = threadIdx.x >> 3, j = threadIdx.x & 7;
    float mx = -3.4e38f;
    for (int ci = j; ci < Lseq; ci += 8) mx = fmaxf(mx, sc[r][ci]);
    part[r][j] = mx; __syncthreads();
    if (j == 0) {
        float m2 = part[r][0];
        for (int q = 1; q < 8; ++q) m2 = fmaxf(m2, part[r][q]);
        part[r][0] = m2;
    }
    __syncthreads();
    float rowmax = part[r][0];
    __syncthreads();
    float sum = 0.f;
    for (int ci = j; ci < Lseq; ci += 8) {
        float e = __expf(sc[r][ci] - rowmax);
        sc[r][ci] = e; sum += e;
    }
    part[r][j] = sum; __syncthreads();
    if (j == 0) {
        float s2 = 0.f;
        for (int q = 0; q < 8; ++q) s2 += part[r][q];
        part[r][0] = s2;
    }
    __syncthreads();
    float inv = 1.f / part[r][0];
    for (int ci = j; ci < Lseq; ci += 8) sc[r][ci] *= inv;
    __syncthreads();

    // ---- Stage 3: O = P * V, wave w handles dh-subtile w; packed output ----
    v8f c = {};
    for (int k0 = 0; k0 < Lseq; k0 += 32) {
        FragBF a, bm;
#pragma unroll
        for (int v = 0; v < 8; ++v) {
            int kk = k0 + ((v >> 2) << 4) + (half << 3) + ((v & 3) << 1);
            a.us[2 * v]     = f2bf_bits(sc[l15][kk]);
            a.us[2 * v + 1] = f2bf_bits(sc[l15][kk + 1]);
            bm.us[2 * v]     = base[(size_t)kk * D3       + 2 * Dm + h * DH + wave * 16 + l15];
            bm.us[2 * v + 1] = base[(size_t)(kk + 1) * D3 + 2 * Dm + h * DH + wave * 16 + l15];
        }
        c = wmma_bf16(a, bm, c);
    }
#pragma unroll
    for (int rr = 0; rr < 8; ++rr) {
        int row = b * Lseq + qt * 16 + rr + half * 8;
        int col = h * DH + wave * 16 + l15;
        out[pack_off(row, col, Dm)] = f2bf_bits(c[rr]);
    }
}

// ---------------------------------------------------------------------------
// Reductions + combine
// ---------------------------------------------------------------------------
__global__ __launch_bounds__(256) void zero_kernel(float* __restrict__ accum,
                                                   float* __restrict__ scal) {
    size_t i = (size_t)blockIdx.x * 256 + threadIdx.x;
    accum[i] = 0.f;
    if (blockIdx.x == 0 && threadIdx.x < 8) scal[threadIdx.x] = 0.f;
}

__global__ __launch_bounds__(256) void add_kernel(float* __restrict__ acc,
                                                  const float* __restrict__ v) {
    size_t i = (size_t)blockIdx.x * 256 + threadIdx.x;
    acc[i] += v[i];
}

__global__ __launch_bounds__(256) void colmean_kernel(const float* __restrict__ in,
                                                      float* __restrict__ outv) {
    __shared__ float red[256];
    int d = blockIdx.x, t = threadIdx.x;
    float s = 0.f;
    for (int r = t; r < Rows; r += 256) s += in[(size_t)r * Dm + d];
    red[t] = s; __syncthreads();
    for (int k = 128; k > 0; k >>= 1) { if (t < k) red[t] += red[t + k]; __syncthreads(); }
    if (t == 0) outv[d] = red[0] * (1.f / Rows);
}

__global__ __launch_bounds__(256) void ortho_kernel(const float* __restrict__ pm,
                                                    float* __restrict__ scal) {
    __shared__ float red[256];
    __shared__ float dots[16];
    int t = threadIdx.x;
    for (int p = 0; p < 16; ++p) {
        int i = p >> 2, j = p & 3;
        float s = 0.f;
        for (int d = t; d < Dm; d += 256) s += pm[i * Dm + d] * pm[j * Dm + d];
        red[t] = s; __syncthreads();
        for (int k = 128; k > 0; k >>= 1) { if (t < k) red[t] += red[t + k]; __syncthreads(); }
        if (t == 0) dots[p] = red[0];
        __syncthreads();
    }
    if (t == 0) {
        float nn[4] = { sqrtf(dots[0]), sqrtf(dots[5]), sqrtf(dots[10]), sqrtf(dots[15]) };
        float acc = 0.f;
        for (int i = 0; i < 4; ++i)
            for (int j = 0; j < 4; ++j) {
                float g = dots[i * 4 + j] / (nn[i] * nn[j]);
                float d2 = g - (i == j ? 1.f : 0.f);
                acc += d2 * d2;
            }
        scal[0] = 1.f - acc * (1.f / 16.f);      // ortho
    }
}

__global__ __launch_bounds__(256) void varpart_kernel(const float* __restrict__ x,
                                                      const float* __restrict__ accum,
                                                      float* __restrict__ scal) {
    __shared__ float r1[256], r2[256];
    int t = threadIdx.x;
    size_t i0 = (size_t)blockIdx.x * 2048 + t;
    float s = 0.f, s2 = 0.f;
    for (int k = 0; k < 8; ++k) {
        size_t i = i0 + (size_t)k * 256;
        float d = x[i] - accum[i] * 0.25f;       // recon = mean of 4 primaries
        s += d; s2 += d * d;
    }
    r1[t] = s; r2[t] = s2; __syncthreads();
    for (int k = 128; k > 0; k >>= 1) {
        if (t < k) { r1[t] += r1[t + k]; r2[t] += r2[t + k]; }
        __syncthreads();
    }
    if (t == 0) { atomicAdd(&scal[1], r1[0]); atomicAdd(&scal[2], r2[0]); }
}

__global__ void rate_kernel(float* __restrict__ scal) {
    double n = (double)NElem;
    double sum = (double)scal[1], sq = (double)scal[2];
    float uvar = (float)((sq - sum * sum / n) / (n - 1.0));   // ddof=1
    float sat = scal[0] * uvar;
    scal[3] = (sat > 0.01f) ? 0.01f : 0.f;                    // ghost-0 rate
}

__global__ __launch_bounds__(256) void final_kernel(const float* __restrict__ accum,
                                                    const float* __restrict__ ghost,
                                                    const float* __restrict__ scal,
                                                    float* __restrict__ outp) {
    size_t i = (size_t)blockIdx.x * 256 + threadIdx.x;
    outp[i] = (accum[i] + scal[3] * ghost[i]) * (1.f / 6.f);  // /(E+G)
}

// ---------------------------------------------------------------------------
// Host-side expert pipeline
// ---------------------------------------------------------------------------
struct Wksp {
    float *accum, *x1, *exp_out, *pm, *scal;
    unsigned short *xn, *hb, *qkv, *attn, *ffn, *wbf;
};

static void run_expert(const float* x, const float* const p[12], int e,
                       const size_t str[12], const Wksp& w, hipStream_t stream) {
    const float* ln1_s = p[0]  + e * str[0];
    const float* ln1_b = p[1]  + e * str[1];
    const float* wqkv  = p[2]  + e * str[2];
    const float* bqkv  = p[3]  + e * str[3];
    const float* wo    = p[4]  + e * str[4];
    const float* bo    = p[5]  + e * str[5];
    const float* ln2_s = p[6]  + e * str[6];
    const float* ln2_b = p[7]  + e * str[7];
    const float* w1    = p[8]  + e * str[8];
    const float* b1    = p[9]  + e * str[9];
    const float* w2    = p[10] + e * str[10];
    const float* b2    = p[11] + e * str[11];

    // xn = LN1(x) (packed A)
    ln_kernel<<<Rows, 256, 0, stream>>>(x, ln1_s, ln1_b, w.xn);
    // qkv = xn @ wqkv + bqkv  (row-major bf16 for attention)
    cvt_pack_kernel<<<(Dm * 3 * Dm) / 256, 256, 0, stream>>>(wqkv, w.wbf, Dm, 3 * Dm);
    gemm_bf16<<<(Rows / 16) * (3 * Dm / 64) / 8, 256, 0, stream>>>(
        (const FragBF*)w.xn, (const FragBF*)w.wbf, bqkv, nullptr, w.qkv,
        Rows, 3 * Dm, Dm, MODE_ROW_BF16);
    // attention (packed A output)
    attn_kernel<<<Bsz * Hn * (Lseq / 16), 128, 0, stream>>>(w.qkv, w.attn);
    // x1 = x + attn @ wo + bo
    cvt_pack_kernel<<<(Dm * Dm) / 256, 256, 0, stream>>>(wo, w.wbf, Dm, Dm);
    gemm_bf16<<<(Rows / 16) * (Dm / 64) / 8, 256, 0, stream>>>(
        (const FragBF*)w.attn, (const FragBF*)w.wbf, bo, x, w.x1,
        Rows, Dm, Dm, MODE_RESID);
    // hb = LN2(x1) (packed A)
    ln_kernel<<<Rows, 256, 0, stream>>>(w.x1, ln2_s, ln2_b, w.hb);
    // ffn = gelu(hb @ w1 + b1) (packed A with K = 2048)
    cvt_pack_kernel<<<(Dm * 4 * Dm) / 256, 256, 0, stream>>>(w1, w.wbf, Dm, 4 * Dm);
    gemm_bf16<<<(Rows / 16) * (4 * Dm / 64) / 8, 256, 0, stream>>>(
        (const FragBF*)w.hb, (const FragBF*)w.wbf, b1, nullptr, w.ffn,
        Rows, 4 * Dm, Dm, MODE_GELU_PACK);
    // exp_out = x1 + ffn @ w2 + b2
    cvt_pack_kernel<<<(4 * Dm * Dm) / 256, 256, 0, stream>>>(w2, w.wbf, 4 * Dm, Dm);
    gemm_bf16<<<(Rows / 16) * (Dm / 64) / 8, 256, 0, stream>>>(
        (const FragBF*)w.ffn, (const FragBF*)w.wbf, b2, w.x1, w.exp_out,
        Rows, Dm, 4 * Dm, MODE_RESID);
}

extern "C" void kernel_launch(void* const* d_in, const int* in_sizes, int n_in,
                              void* d_out, int out_size, void* d_ws, size_t ws_size,
                              hipStream_t stream) {
    (void)in_sizes; (void)n_in; (void)out_size; (void)ws_size;

    const float* x = (const float*)d_in[0];
    const float* P[12];  for (int i = 0; i < 12; ++i) P[i]  = (const float*)d_in[2 + i];
    const float* Gp[12]; for (int i = 0; i < 12; ++i) Gp[i] = (const float*)d_in[14 + i];

    static const size_t str[12] = {
        Dm, Dm, (size_t)Dm * 3 * Dm, 3 * Dm, (size_t)Dm * Dm, Dm,
        Dm, Dm, (size_t)Dm * 4 * Dm, 4 * Dm, (size_t)4 * Dm * Dm, Dm };

    const size_t MB = 1u << 20;
    char* ws = (char*)d_ws;
    Wksp w;
    w.accum   = (float*)(ws + 0 * MB);                     // 8 MB
    w.x1      = (float*)(ws + 8 * MB);                     // 8 MB
    w.exp_out = (float*)(ws + 16 * MB);                    // 8 MB
    w.xn      = (unsigned short*)(ws + 24 * MB);           // 4 MB  (packed A)
    w.hb      = (unsigned short*)(ws + 28 * MB);           // 4 MB  (packed A)
    w.qkv     = (unsigned short*)(ws + 32 * MB);           // 12 MB (row-major)
    w.attn    = (unsigned short*)(ws + 44 * MB);           // 4 MB  (packed A)
    w.ffn     = (unsigned short*)(ws + 48 * MB);           // 16 MB (packed A)
    w.wbf     = (unsigned short*)(ws + 64 * MB);           // 2 MB  (packed B, reused)
    w.pm      = (float*)(ws + 66 * MB);                    // 4x512 f32
    w.scal    = (float*)(ws + 66 * MB + 16384);            // [ortho,sum,sumsq,rate]

    zero_kernel<<<NElem / 256, 256, 0, stream>>>(w.accum, w.scal);

    // 4 primary experts: accumulate outputs + per-expert column means
    for (int e = 0; e < 4; ++e) {
        run_expert(x, P, e, str, w, stream);
        add_kernel<<<NElem / 256, 256, 0, stream>>>(w.accum, w.exp_out);
        colmean_kernel<<<Dm, 256, 0, stream>>>(w.exp_out, w.pm + e * Dm);
    }

    // Saturation metrics -> ghost-0 activation rate (device-resident)
    ortho_kernel<<<1, 256, 0, stream>>>(w.pm, w.scal);
    varpart_kernel<<<NElem / 2048, 256, 0, stream>>>(x, w.accum, w.scal);
    rate_kernel<<<1, 1, 0, stream>>>(w.scal);

    // Ghost expert 0 (ghost 1 is always scaled by zero -> skipped, still /6)
    run_expert(x, Gp, 0, str, w, stream);

    final_kernel<<<NElem / 256, 256, 0, stream>>>(w.accum, w.exp_out, w.scal,
                                                  (float*)d_out);
}
